// fuse_v8DetectionLoss_55765855371952
// MI455X (gfx1250) — matrix operand order repaired and checked by reading, not compiled, not run on hardware
//
#include <hip/hip_runtime.h>
#include <math.h>

// ---------------- problem constants (from reference) ----------------
#define B_    16
#define A_    8400            // 6400 + 1600 + 400
#define M_    60
#define NO_   65              // NC + 4*REG_MAX
#define EPSA  1e-9f
#define INV_PI2_4 0.405284734569351f   // 4 / pi^2

typedef __attribute__((ext_vector_type(16))) _Float16 v16h;
typedef __attribute__((ext_vector_type(8)))  float    v8f;

// ---------------- helpers ----------------
__device__ __forceinline__ void anchor_of(int a, float& gx, float& gy, float& st) {
  int i, w;
  if (a < 6400)      { i = a;        w = 80; st = 8.f;  }
  else if (a < 8000) { i = a - 6400; w = 40; st = 16.f; }
  else               { i = a - 8000; w = 20; st = 32.f; }
  gx = (float)(i % w) + 0.5f;
  gy = (float)(i / w) + 0.5f;
}

__device__ __forceinline__ const float* feat_ptr(const float* f0, const float* f1,
                                                 const float* f2, int b, int c, int a) {
  if (a < 6400) return f0 + ((size_t)(b * NO_ + c)) * 6400 + a;
  if (a < 8000) return f1 + ((size_t)(b * NO_ + c)) * 1600 + (a - 6400);
  return f2 + ((size_t)(b * NO_ + c)) * 400 + (a - 8000);
}

// exact CIoU per reference (eps placement matters)
__device__ __forceinline__ float ciou_f(float b1x1, float b1y1, float b1x2, float b1y2,
                                        float b2x1, float b2y1, float b2x2, float b2y2) {
  const float eps = 1e-7f;
  float w1 = b1x2 - b1x1, h1 = b1y2 - b1y1 + eps;
  float w2 = b2x2 - b2x1, h2 = b2y2 - b2y1 + eps;
  float iw = fmaxf(fminf(b1x2, b2x2) - fmaxf(b1x1, b2x1), 0.f);
  float ih = fmaxf(fminf(b1y2, b2y2) - fmaxf(b1y1, b2y1), 0.f);
  float inter = iw * ih;
  float uni = w1 * h1 + w2 * h2 - inter + eps;
  float iou = inter / uni;
  float cw = fmaxf(b1x2, b2x2) - fminf(b1x1, b2x1);
  float ch = fmaxf(b1y2, b2y2) - fminf(b1y1, b2y1);
  float c2 = cw * cw + ch * ch + eps;
  float dx = (b2x1 + b2x2 - b1x1 - b1x2);
  float dy = (b2y1 + b2y2 - b1y1 - b1y2);
  float rho2 = (dx * dx + dy * dy) * 0.25f;
  float dat = atanf(w2 / h2) - atanf(w1 / h1);
  float v = INV_PI2_4 * dat * dat;
  float alpha = v / (v - iou + 1.f + eps);
  return iou - (rho2 / c2 + v * alpha);
}

// ---------------- K1: DFL decode via WMMA ----------------
// One wave32 per 16 anchors. Lane l owns {anchor j=l&15, K-half kh=l>>4}:
// it loads+softmaxes one full 16-bin side, which matches the contiguous K order (K = 16h + s) this code uses
// for a 16-bit 32x16 B matrix (lanes 0-15: K=0..15, lanes 16-31: K=16..31,
// column = lane%16). A-operand carries proj in rows 0/1 (row0: K<16, row1: K>=16)
// in the same contiguous K order as B. D rows 0/1 land in d[0],d[1] of lanes 0-15.
__global__ __launch_bounds__(32) void k_decode(const float* __restrict__ f0,
                                               const float* __restrict__ f1,
                                               const float* __restrict__ f2,
                                               float* __restrict__ pred_bboxes,
                                               float* __restrict__ sig_scores) {
  int wave = blockIdx.x;                 // B_ * (A_/16) waves
  int b = wave / (A_ / 16);
  int abase = (wave % (A_ / 16)) * 16;
  int l  = threadIdx.x;
  int j  = l & 15;
  int kh = l >> 4;
  int a  = abase + j;

  // Build A operand (projection matrix) in contiguous K order (K = 16*kh + e).
  v16h avec;
  int g = kh ? 8 : 0;
  int Mrow = j;                          // lanes 0-15 -> M=lane, 16-31 -> M=lane-16
#pragma unroll
  for (int e = 0; e < 16; ++e) {
    int v = e >> 1, h = e & 1;
    int K = 16 * kh + e; (void)v; (void)h; (void)g;
    float val = 0.f;
    if (Mrow == 0 && K < 16)  val = (float)K;         // proj[K]
    if (Mrow == 1 && K >= 16) val = (float)(K - 16);  // proj[K-16]
    avec[e] = (_Float16)val;
  }

  v8f acc0 = {0.f, 0.f, 0.f, 0.f, 0.f, 0.f, 0.f, 0.f};
  v8f acc1 = acc0;
  float lg[16];

  // WMMA #1: sides {0,1}; this lane's side = kh
#pragma unroll
  for (int k = 0; k < 16; ++k) lg[k] = *feat_ptr(f0, f1, f2, b, kh * 16 + k, a);
  {
    float mx = lg[0];
#pragma unroll
    for (int k = 1; k < 16; ++k) mx = fmaxf(mx, lg[k]);
    float e[16], s = 0.f;
#pragma unroll
    for (int k = 0; k < 16; ++k) { e[k] = __expf(lg[k] - mx); s += e[k]; }
    float inv = 1.f / s;
    v16h bvec;
#pragma unroll
    for (int k = 0; k < 16; ++k) bvec[k] = (_Float16)(e[k] * inv);
    acc0 = __builtin_amdgcn_wmma_f32_16x16x32_f16(false, avec, false, bvec,
                                                  (short)0, acc0, false, false);
  }
  // WMMA #2: sides {2,3}; this lane's side = 2+kh
#pragma unroll
  for (int k = 0; k < 16; ++k) lg[k] = *feat_ptr(f0, f1, f2, b, (2 + kh) * 16 + k, a);
  {
    float mx = lg[0];
#pragma unroll
    for (int k = 1; k < 16; ++k) mx = fmaxf(mx, lg[k]);
    float e[16], s = 0.f;
#pragma unroll
    for (int k = 0; k < 16; ++k) { e[k] = __expf(lg[k] - mx); s += e[k]; }
    float inv = 1.f / s;
    v16h bvec;
#pragma unroll
    for (int k = 0; k < 16; ++k) bvec[k] = (_Float16)(e[k] * inv);
    acc1 = __builtin_amdgcn_wmma_f32_16x16x32_f16(false, avec, false, bvec,
                                                  (short)0, acc1, false, false);
  }

  if (l < 16) {                          // rows M=0,1 live in lanes 0-15, VGPR 0/1
    float d0 = acc0[0], d1 = acc0[1], d2 = acc1[0], d3 = acc1[1];
    float gx, gy, st;
    anchor_of(a, gx, gy, st);
    float4 bb = make_float4(gx - d0, gy - d1, gx + d2, gy + d3);   // grid units
    ((float4*)pred_bboxes)[(size_t)b * A_ + a] = bb;
    float x = *feat_ptr(f0, f1, f2, b, 64, a);
    sig_scores[(size_t)b * A_ + a] = 1.f / (1.f + __expf(-x));
  }
}

// ---------------- K2: per-(b,m) row: CIoU, metric, faithful top-10 ----------------
__global__ __launch_bounds__(256) void k_assign_rows(const float* __restrict__ gt_bboxes,
                                                     const float* __restrict__ pred_bboxes,
                                                     const float* __restrict__ sig,
                                                     float* __restrict__ overlaps,
                                                     unsigned char* __restrict__ mask_pos) {
  __shared__ float sval[256 * 10];
  __shared__ int   sidx[256 * 10];
  __shared__ int   head[256];
  int bm = blockIdx.x;
  int b = bm / M_;
  const float* gt = gt_bboxes + (size_t)bm * 4;
  float gx1 = gt[0], gy1 = gt[1], gx2 = gt[2], gy2 = gt[3];
  float mask_gt = ((gx1 + gy1 + gx2 + gy2) > 0.f) ? 1.f : 0.f;
  int tid = threadIdx.x;
  float tv[10]; int ti[10];
#pragma unroll
  for (int k = 0; k < 10; ++k) { tv[k] = -1.f; ti[k] = 0x7fffffff; }
  size_t row = (size_t)bm * A_;
  for (int a = tid; a < A_; a += 256) {
    if (a + 1024 < A_)
      __builtin_prefetch((const void*)(pred_bboxes + ((size_t)b * A_ + a + 1024) * 4), 0, 0);
    float ax, ay, st;
    anchor_of(a, ax, ay, st);
    float px = ax * st, py = ay * st;                     // anchor in pixels
    float dmin = fminf(fminf(px - gx1, py - gy1), fminf(gx2 - px, gy2 - py));
    float ing = (dmin > EPSA) ? 1.f : 0.f;
    float mask = ing * mask_gt;
    float4 pb = ((const float4*)pred_bboxes)[(size_t)b * A_ + a];
    float ov = 0.f;
    if (mask > 0.f) {
      float c = ciou_f(gx1, gy1, gx2, gy2, pb.x * st, pb.y * st, pb.z * st, pb.w * st);
      ov = fmaxf(c, 0.f);
    }
    overlaps[row + a] = ov;
    mask_pos[row + a] = 0;
    float bs = sig[(size_t)b * A_ + a] * mask;
    float o3 = ov * ov * ov;
    float metric = sqrtf(bs) * (o3 * o3) * ing;           // align * mask_in_gts
    if (metric > tv[9]) {                                  // desc sort, ties keep earlier a
      int p = 9;
      while (p > 0 && metric > tv[p - 1]) { tv[p] = tv[p - 1]; ti[p] = ti[p - 1]; --p; }
      tv[p] = metric; ti[p] = a;
    }
  }
#pragma unroll
  for (int k = 0; k < 10; ++k) { sval[tid * 10 + k] = tv[k]; sidx[tid * 10 + k] = ti[k]; }
  head[tid] = tid * 10;
  __syncthreads();
  if (tid == 0) {
    // 10-way merge of 256 sorted lists; ties -> lowest anchor index (= jax top_k)
    for (int sel = 0; sel < 10; ++sel) {
      float bv = -2.f; int bi = 0x7fffffff; int bt = 0;
      for (int t = 0; t < 256; ++t) {
        int h = head[t];
        if (h < t * 10 + 10) {
          float v = sval[h]; int ix = sidx[h];
          if (v > bv || (v == bv && ix < bi)) { bv = v; bi = ix; bt = t; }
        }
      }
      head[bt]++;
      if (mask_gt > 0.f && bi < A_) {
        float ax, ay, st;
        anchor_of(bi, ax, ay, st);
        float px = ax * st, py = ay * st;
        float dmin = fminf(fminf(px - gx1, py - gy1), fminf(gx2 - px, gy2 - py));
        if (dmin > EPSA) mask_pos[row + bi] = 1;          // * mask_in_gts * mask_gt
      }
    }
  }
}

// ---------------- K3: per-(b,a) column resolve ----------------
__global__ __launch_bounds__(256) void k_resolve(const float* __restrict__ overlaps,
                                                 unsigned char* __restrict__ mask_pos,
                                                 float* __restrict__ fg_mask,
                                                 int* __restrict__ tgi) {
  int idx = blockIdx.x * 256 + threadIdx.x;
  if (idx >= B_ * A_) return;
  int b = idx / A_, a = idx - b * A_;
  size_t col = (size_t)b * M_ * A_ + a;
  int fg = 0;
  for (int m = 0; m < M_; ++m) fg += mask_pos[col + (size_t)m * A_];
  int t = 0;
  if (fg > 1) {
    float best = -1.f; int bmx = 0;
    for (int m = 0; m < M_; ++m) {
      float o = overlaps[col + (size_t)m * A_];
      if (o > best) { best = o; bmx = m; }                // first max, like argmax
    }
    for (int m = 0; m < M_; ++m) mask_pos[col + (size_t)m * A_] = (m == bmx) ? 1 : 0;
    fg = 1; t = bmx;
  } else if (fg == 1) {
    for (int m = 0; m < M_; ++m)
      if (mask_pos[col + (size_t)m * A_]) { t = m; break; }
  }
  fg_mask[idx] = (float)fg;
  tgi[idx] = t;
}

// ---------------- K4: per-(b,m) row maxes (align recomputed from overlaps) ------
__global__ __launch_bounds__(256) void k_row_maxes(const float* __restrict__ gt_bboxes,
                                                   const float* __restrict__ sig,
                                                   const float* __restrict__ overlaps,
                                                   const unsigned char* __restrict__ mask_pos,
                                                   float* __restrict__ pos_align,
                                                   float* __restrict__ pos_ov) {
  __shared__ float sa[256], so[256];
  int bm = blockIdx.x;
  int b = bm / M_;
  const float* gt = gt_bboxes + (size_t)bm * 4;
  float gx1 = gt[0], gy1 = gt[1], gx2 = gt[2], gy2 = gt[3];
  float mask_gt = ((gx1 + gy1 + gx2 + gy2) > 0.f) ? 1.f : 0.f;
  float ma = 0.f, mo = 0.f;
  size_t row = (size_t)bm * A_;
  for (int a = threadIdx.x; a < A_; a += 256) {
    if (mask_pos[row + a]) {
      float ov = overlaps[row + a];
      float ax, ay, st;
      anchor_of(a, ax, ay, st);
      float px = ax * st, py = ay * st;
      float dmin = fminf(fminf(px - gx1, py - gy1), fminf(gx2 - px, gy2 - py));
      float ing = (dmin > EPSA) ? 1.f : 0.f;
      float bs = sig[(size_t)b * A_ + a] * ing * mask_gt;
      float o3 = ov * ov * ov;
      float al = sqrtf(bs) * (o3 * o3);
      ma = fmaxf(ma, al);
      mo = fmaxf(mo, ov);
    }
  }
  int tid = threadIdx.x;
  sa[tid] = ma; so[tid] = mo;
  __syncthreads();
  for (int off = 128; off > 0; off >>= 1) {
    if (tid < off) { sa[tid] = fmaxf(sa[tid], sa[tid + off]); so[tid] = fmaxf(so[tid], so[tid + off]); }
    __syncthreads();
  }
  if (tid == 0) { pos_align[bm] = sa[0]; pos_ov[bm] = so[0]; }
}

// ---------------- K5: fused loss pass ----------------
__global__ __launch_bounds__(256) void k_loss(const float* __restrict__ f0,
                                              const float* __restrict__ f1,
                                              const float* __restrict__ f2,
                                              const float* __restrict__ gt_bboxes,
                                              const float* __restrict__ pred_bboxes,
                                              const float* __restrict__ sig,
                                              const float* __restrict__ overlaps,
                                              const unsigned char* __restrict__ mask_pos,
                                              const float* __restrict__ pos_align,
                                              const float* __restrict__ pos_ov,
                                              const float* __restrict__ fg_mask,
                                              const int* __restrict__ tgi,
                                              float* __restrict__ acc) {
  __shared__ float red[4][256];
  int idx = blockIdx.x * 256 + threadIdx.x;
  float ts = 0.f, bce = 0.f, li = 0.f, ld = 0.f;
  if (idx < B_ * A_) {
    int b = idx / A_, a = idx - b * A_;
    float ax, ay, st;
    anchor_of(a, ax, ay, st);
    // norm[b,a] = max_m align*mask_pos * pos_ov / (pos_align + eps)
    float norm = 0.f;
    size_t col = (size_t)b * M_ * A_ + a;
    for (int m = 0; m < M_; ++m) {
      if (mask_pos[col + (size_t)m * A_]) {
        const float* gt = gt_bboxes + ((size_t)b * M_ + m) * 4;
        float mask_gt = ((gt[0] + gt[1] + gt[2] + gt[3]) > 0.f) ? 1.f : 0.f;
        float px = ax * st, py = ay * st;
        float dmin = fminf(fminf(px - gt[0], py - gt[1]), fminf(gt[2] - px, gt[3] - py));
        float ing = (dmin > EPSA) ? 1.f : 0.f;
        float ov = overlaps[col + (size_t)m * A_];
        float bs = sig[idx] * ing * mask_gt;
        float o3 = ov * ov * ov;
        float al = sqrtf(bs) * (o3 * o3);
        int bm = b * M_ + m;
        norm = fmaxf(norm, al * pos_ov[bm] / (pos_align[bm] + EPSA));
      }
    }
    float fg = fg_mask[idx];
    ts = (fg > 0.f) ? norm : 0.f;
    // BCE with logits
    float x = *feat_ptr(f0, f1, f2, b, 64, a);
    bce = fmaxf(x, 0.f) - x * ts + log1pf(__expf(-fabsf(x)));
    float w = ts * fg;
    if (w > 0.f) {
      int t = tgi[idx];
      const float* gt = gt_bboxes + ((size_t)b * M_ + t) * 4;
      float inv = 1.f / st;
      float tx1 = gt[0] * inv, ty1 = gt[1] * inv, tx2 = gt[2] * inv, ty2 = gt[3] * inv;
      float4 pb = ((const float4*)pred_bboxes)[idx];
      float iou = ciou_f(pb.x, pb.y, pb.z, pb.w, tx1, ty1, tx2, ty2);
      li = (1.f - iou) * w;
      float t4[4] = {ax - tx1, ay - ty1, tx2 - ax, ty2 - ay};
      float df = 0.f;
#pragma unroll
      for (int s = 0; s < 4; ++s) {
        float v = fminf(fmaxf(t4[s], 0.f), 14.99f);
        int tl = (int)floorf(v);
        int tr = (tl + 1 < 15) ? tl + 1 : 15;
        float wl = (float)(tl + 1) - v, wr = 1.f - wl;
        float lg[16]; float mx = -1e30f;
#pragma unroll
        for (int k = 0; k < 16; ++k) { lg[k] = *feat_ptr(f0, f1, f2, b, s * 16 + k, a); mx = fmaxf(mx, lg[k]); }
        float se = 0.f;
#pragma unroll
        for (int k = 0; k < 16; ++k) se += __expf(lg[k] - mx);
        float lse = logf(se);
        df += -((lg[tl] - mx - lse) * wl + (lg[tr] - mx - lse) * wr);
      }
      ld = df * 0.25f * w;
    }
  }
  int tid = threadIdx.x;
  red[0][tid] = ts; red[1][tid] = bce; red[2][tid] = li; red[3][tid] = ld;
  __syncthreads();
  for (int off = 128; off > 0; off >>= 1) {
    if (tid < off) {
#pragma unroll
      for (int q = 0; q < 4; ++q) red[q][tid] += red[q][tid + off];
    }
    __syncthreads();
  }
  if (tid == 0) {
    atomicAdd(&acc[0], red[0][0]);
    atomicAdd(&acc[1], red[1][0]);
    atomicAdd(&acc[2], red[2][0]);
    atomicAdd(&acc[3], red[3][0]);
  }
}

// ---------------- K6: finalize ----------------
__global__ void k_final(const float* __restrict__ acc, float* __restrict__ out) {
  float tss = fmaxf(acc[0], 1.f);
  float li = acc[2] / tss * 7.5f;   // HYP_BOX
  float lc = acc[1] / tss * 0.5f;   // HYP_CLS
  float ldf = acc[3] / tss * 1.5f;  // HYP_DFL
  out[0] = (li + lc + ldf) * (float)B_;
  out[1] = li;
  out[2] = lc;
  out[3] = ldf;
}

// ---------------- launcher ----------------
extern "C" void kernel_launch(void* const* d_in, const int* in_sizes, int n_in,
                              void* d_out, int out_size, void* d_ws, size_t ws_size,
                              hipStream_t stream) {
  (void)in_sizes; (void)n_in; (void)out_size; (void)ws_size;
  const float* f0  = (const float*)d_in[0];
  const float* f1  = (const float*)d_in[1];
  const float* f2  = (const float*)d_in[2];
  // d_in[3] = gt_labels (int) unused: NC==1 -> labels clip to 0
  const float* gtb = (const float*)d_in[4];
  float* out = (float*)d_out;

  char* p = (char*)d_ws;
  auto carve = [&](size_t bytes) -> char* {
    char* r = p;
    p += (bytes + 255) & ~(size_t)255;
    return r;
  };
  float*         pred_bboxes = (float*)carve((size_t)B_ * A_ * 4 * sizeof(float));
  float*         sig         = (float*)carve((size_t)B_ * A_ * sizeof(float));
  float*         overlaps    = (float*)carve((size_t)B_ * M_ * A_ * sizeof(float));
  unsigned char* mask_pos    = (unsigned char*)carve((size_t)B_ * M_ * A_);
  float*         pos_align   = (float*)carve((size_t)B_ * M_ * sizeof(float));
  float*         pos_ov      = (float*)carve((size_t)B_ * M_ * sizeof(float));
  float*         fg          = (float*)carve((size_t)B_ * A_ * sizeof(float));
  int*           tgi         = (int*)carve((size_t)B_ * A_ * sizeof(int));
  float*         acc         = (float*)carve(4 * sizeof(float));

  hipMemsetAsync(acc, 0, 4 * sizeof(float), stream);

  k_decode<<<B_ * (A_ / 16), 32, 0, stream>>>(f0, f1, f2, pred_bboxes, sig);
  k_assign_rows<<<B_ * M_, 256, 0, stream>>>(gtb, pred_bboxes, sig, overlaps, mask_pos);
  k_resolve<<<(B_ * A_ + 255) / 256, 256, 0, stream>>>(overlaps, mask_pos, fg, tgi);
  k_row_maxes<<<B_ * M_, 256, 0, stream>>>(gtb, sig, overlaps, mask_pos, pos_align, pos_ov);
  k_loss<<<(B_ * A_ + 255) / 256, 256, 0, stream>>>(f0, f1, f2, gtb, pred_bboxes, sig,
                                                    overlaps, mask_pos, pos_align, pos_ov,
                                                    fg, tgi, acc);
  k_final<<<1, 1, 0, stream>>>(acc, out);
}
